// PairEmbedding_43585328120334
// MI455X (gfx1250) — matrix operand order repaired and checked by reading, not compiled, not run on hardware
//
#include <hip/hip_runtime.h>
#include <hip/hip_bf16.h>

typedef __attribute__((ext_vector_type(16))) _Float16 v16h;
typedef __attribute__((ext_vector_type(8)))  _Float16 v8h;
typedef __attribute__((ext_vector_type(8)))  float    v8f;

// ---------------- LDS layout (bytes) ----------------
// [0,2048)        : float tables: f_az[64], f_po[64], means[128], stdInv[128], coef[128]
// [2048,137216)   : dir_W^T  f16 [256 cols][264]   (K=256 padded to 264)
// [137216,172032) : proj1_W^T f16 [128 cols][136]  (K=128 padded to 136)
// [172032,241664) : proj2_W^T f16 [256 cols][136]
// [241664,309248) : per-wave stage: 8 waves x [16 rows][264] halfs
#define SMEM_BYTES 309248
#define DIRW_OFF   2048
#define P1_OFF     137216
#define P2_OFF     172032
#define STAGE_OFF  241664

// A-matrix 16x32 f16 fragment from row-major LDS panel (stride in halfs).
// ISA 7.12.2: lanes 0-15 row=lane, elems 0..7 = K(kb..kb+7), 8..15 = K(kb+16..kb+23);
// lanes 16-31 row=lane-16, K(kb+8..15) and K(kb+24..31).
__device__ __forceinline__ v16h load_frag_A(const _Float16* base, int strideH,
                                            int kbase, int lane) {
  int r = lane & 15, hi = lane >> 4;
  const _Float16* p = base + r * strideH + kbase + 8 * hi;
  v8h lo = *(const v8h*)p;
  v8h hh = *(const v8h*)(p + 16);
  v16h out;
#pragma unroll
  for (int e = 0; e < 8; ++e) { out[e] = lo[e]; out[e + 8] = hh[e]; }
  return out;
}

// B-matrix 32x16 f16 fragment from column-major LDS panel (stride in halfs).
// Column = nbase + lane%16; lanes 0-15 hold K kb..kb+7, kb+16..kb+23; lanes 16-31 K kb+8..kb+15, kb+24..kb+31.
__device__ __forceinline__ v16h load_frag_B(const _Float16* base, int strideH,
                                            int nbase, int kbase, int lane) {
  int c = lane & 15, hi = lane >> 4;
  const _Float16* p = base + (nbase + c) * strideH + kbase + 8 * hi;
  v8h lo = *(const v8h*)p;
  v8h hh = *(const v8h*)(p + 16);
  v16h out;
#pragma unroll
  for (int e = 0; e < 8; ++e) { out[e] = lo[e]; out[e + 8] = hh[e]; }
  return out;
}

__device__ __forceinline__ v8f wmma_f16(v16h a, v16h b, v8f c) {
  return __builtin_amdgcn_wmma_f32_16x16x32_f16(false, a, false, b, (short)0, c,
                                                false, false);
}

// ---------------- node = emb_z[z] + (ec[z]@config_W + config_b | 0) + mult_W[1] + charge_W[0]
__global__ void __launch_bounds__(256) node_kernel(
    const int* __restrict__ an, const float* __restrict__ emb_z,
    const float* __restrict__ ec, const float* __restrict__ cW,
    const float* __restrict__ cb, const float* __restrict__ multW,
    const float* __restrict__ chargeW, float* __restrict__ node_out) {
  int bi = blockIdx.x;            // b*256 + i
  int b = bi >> 8, ii = bi & 255;
  int d = threadIdx.x;
  int zv = (ii == 0) ? 101 : an[b * 255 + ii - 1];
  float acc = emb_z[zv * 256 + d] + multW[256 + d] + chargeW[d];
  if (ii != 0) {
    float s = cb[d];
    const float* e = ec + zv * 20;
#pragma unroll
    for (int f = 0; f < 20; ++f) s += e[f] * cW[f * 256 + d];
    acc += s;
  }
  node_out[(size_t)bi * 256 + d] = acc;
}

// ---------------- fused pair kernel: sinusoids->dir GEMM, gauss->proj1->GELU->proj2
__global__ void __launch_bounds__(256) pair_kernel(
    const float* __restrict__ positions, const int* __restrict__ an,
    const float* __restrict__ gbf_means, const float* __restrict__ gbf_stds,
    const float* __restrict__ gbf_mul, const float* __restrict__ gbf_bias,
    const float* __restrict__ proj1_W, const float* __restrict__ proj1_b,
    const float* __restrict__ proj2_W, const float* __restrict__ proj2_b,
    const float* __restrict__ dir_W, const float* __restrict__ dir_b,
    float* __restrict__ gbf_out, float* __restrict__ dir_out) {
  extern __shared__ char smem[];
  float* fAz    = (float*)smem;
  float* fPo    = fAz + 64;
  float* meansL = fPo + 64;
  float* stdInv = meansL + 128;
  float* coefL  = stdInv + 128;
  _Float16* dirWT    = (_Float16*)(smem + DIRW_OFF);
  _Float16* p1T      = (_Float16*)(smem + P1_OFF);
  _Float16* p2T      = (_Float16*)(smem + P2_OFF);
  _Float16* stageAll = (_Float16*)(smem + STAGE_OFF);

  const int tid = threadIdx.x;
  // angular frequency tables: f = 2pi * exp(-(ln(2e-4) + k*step))
  if (tid < 64) {
    float k = (float)tid;
    fAz[tid] = 6.2831853071f * __expf(8.5171931914f - k * 0.1753685308f);
  } else if (tid < 128) {
    float k = (float)(tid - 64);
    fPo[tid - 64] = 6.2831853071f * __expf(8.5171931914f - k * 0.1643661946f);
  } else if (tid < 256) {
    int k = tid - 128;
    meansL[k] = gbf_means[k];
    float s = fabsf(gbf_stds[k]) + 0.01f;
    stdInv[k] = 1.0f / s;
    coefL[k]  = 0.3989424269f / s;   // 1/sqrt(2*3.14159)
  }
  // weight panels, transposed to column-major f16
  for (int idx = tid; idx < 256 * 256; idx += 256) {
    int k = idx >> 8, n = idx & 255;
    dirWT[n * 264 + k] = (_Float16)dir_W[idx];
  }
  for (int idx = tid; idx < 128 * 128; idx += 256) {
    int k = idx >> 7, n = idx & 127;
    p1T[n * 136 + k] = (_Float16)proj1_W[idx];
  }
  for (int idx = tid; idx < 128 * 256; idx += 256) {
    int k = idx >> 8, n = idx & 255;
    p2T[n * 136 + k] = (_Float16)proj2_W[idx];
  }
  __syncthreads();

  const int lane = tid & 31;
  const int wave = tid >> 5;
  _Float16* stage = stageAll + wave * 4224;  // 16 rows * 264 halfs
  const int waveGlobal = blockIdx.x * 8 + wave;
  const int numWaves   = gridDim.x * 8;

  for (int tile = waveGlobal; tile < 16384; tile += numWaves) {
    const int bi    = tile >> 4;           // b*256 + i
    const int b     = bi >> 8;
    const int i     = bi & 255;
    const int jbase = (tile & 15) << 4;
    const int l     = lane & 15;
    const int j     = jbase + l;

    // ---- per-pair geometry (lanes 0-15 and 16-31 duplicate; no divergence) ----
    float pix = 0.f, piy = 0.f, piz = 0.f, pjx = 0.f, pjy = 0.f, pjz = 0.f;
    if (i > 0) {
      const float* p = positions + ((size_t)b * 255 + (i - 1)) * 3;
      pix = p[0]; piy = p[1]; piz = p[2];
    }
    if (j > 0) {
      const float* p = positions + ((size_t)b * 255 + (j - 1)) * 3;
      pjx = p[0]; pjy = p[1]; pjz = p[2];
    }
    float dx = pjx - pix, dy = pjy - piy, dz = pjz - piz;
    float sq   = dx * dx + dy * dy + dz * dz;
    float dist = sqrtf(sq + 1e-12f);
    float inv  = 1.0f / (dist + 1e-5f);
    float ndy = dy * inv, ndz = dz * inv, ndx = dx * inv;
    float az = (i == j) ? 0.0f : atan2f(ndy, ndx);
    float cz = fminf(fmaxf(ndz, -1.0f + 1e-6f), 1.0f - 1e-6f);
    float po = acosf(cz);
    int zi = (i == 0) ? 101 : an[(size_t)b * 255 + (i - 1)];
    int zj = (j == 0) ? 101 : an[(size_t)b * 255 + (j - 1)];
    float mul = gbf_mul[zi] + gbf_mul[zj + 128];
    float bia = gbf_bias[zi] + gbf_bias[zj + 128];
    float gx  = mul * dist + bia;

    const size_t mbase = (size_t)tile * 16;  // global pair-row base
    const int hi = lane >> 4, cc = lane & 15;

    // ---- sinusoid A-panel (16 x 256) into stage ----
    for (int r = 0; r < 16; ++r) {
      float azr = __shfl(az, r, 32);
      float por = __shfl(po, r, 32);
#pragma unroll
      for (int part = 0; part < 8; ++part) {
        int k = part * 32 + lane;
        float v;
        if (k < 64)       v = __sinf(azr * fAz[k]);
        else if (k < 128) v = __cosf(azr * fAz[k - 64]);
        else if (k < 192) v = __sinf(por * fPo[k - 128]);
        else              v = __cosf(por * fPo[k - 192]);
        stage[r * 264 + k] = (_Float16)v;
      }
    }

    // ---- dir_out = sinusoids @ dir_W + dir_b  (K=256, N=256) ----
    v16h A[8];
#pragma unroll
    for (int kb = 0; kb < 8; ++kb) A[kb] = load_frag_A(stage, 264, kb * 32, lane);
    for (int nb = 0; nb < 16; ++nb) {
      float b0 = dir_b[nb * 16 + cc];
      v8f c;
#pragma unroll
      for (int e = 0; e < 8; ++e) c[e] = b0;
#pragma unroll
      for (int kb = 0; kb < 8; ++kb)
        c = wmma_f16(A[kb], load_frag_B(dirWT, 264, nb * 16, kb * 32, lane), c);
#pragma unroll
      for (int v = 0; v < 8; ++v)
        dir_out[(mbase + hi * 8 + v) * 256 + nb * 16 + cc] = c[v];
    }

    // ---- gauss A-panel (16 x 128) into stage (overwrites sinusoids) ----
    for (int r = 0; r < 16; ++r) {
      float gxr = __shfl(gx, r, 32);
#pragma unroll
      for (int part = 0; part < 4; ++part) {
        int k = part * 32 + lane;
        float t = (gxr - meansL[k]) * stdInv[k];
        stage[r * 264 + k] = (_Float16)(__expf(-0.5f * t * t) * coefL[k]);
      }
    }

    // ---- h = gelu(gauss @ proj1_W + b1), staged at half-offset 136 ----
    v16h G[4];
#pragma unroll
    for (int kb = 0; kb < 4; ++kb) G[kb] = load_frag_A(stage, 264, kb * 32, lane);
    for (int nb = 0; nb < 8; ++nb) {
      float b0 = proj1_b[nb * 16 + cc];
      v8f c;
#pragma unroll
      for (int e = 0; e < 8; ++e) c[e] = b0;
#pragma unroll
      for (int kb = 0; kb < 4; ++kb)
        c = wmma_f16(G[kb], load_frag_B(p1T, 136, nb * 16, kb * 32, lane), c);
#pragma unroll
      for (int v = 0; v < 8; ++v) {
        float x = c[v];
        float g = 0.5f * x * (1.0f + erff(x * 0.7071067812f));  // exact gelu
        stage[(hi * 8 + v) * 264 + 136 + nb * 16 + cc] = (_Float16)g;
      }
    }

    // ---- gbf_out = h @ proj2_W + b2  (K=128, N=256) ----
    v16h H[4];
#pragma unroll
    for (int kb = 0; kb < 4; ++kb)
      H[kb] = load_frag_A(stage + 136, 264, kb * 32, lane);
    for (int nb = 0; nb < 16; ++nb) {
      float b0 = proj2_b[nb * 16 + cc];
      v8f c;
#pragma unroll
      for (int e = 0; e < 8; ++e) c[e] = b0;
#pragma unroll
      for (int kb = 0; kb < 4; ++kb)
        c = wmma_f16(H[kb], load_frag_B(p2T, 136, nb * 16, kb * 32, lane), c);
#pragma unroll
      for (int v = 0; v < 8; ++v)
        gbf_out[(mbase + hi * 8 + v) * 256 + nb * 16 + cc] = c[v];
    }
  }
}

extern "C" void kernel_launch(void* const* d_in, const int* in_sizes, int n_in,
                              void* d_out, int out_size, void* d_ws, size_t ws_size,
                              hipStream_t stream) {
  (void)in_sizes; (void)n_in; (void)d_ws; (void)ws_size; (void)out_size;
  const float* positions = (const float*)d_in[0];
  const int*   an        = (const int*)d_in[1];
  const float* emb_z     = (const float*)d_in[2];
  const float* ec        = (const float*)d_in[3];
  const float* cW        = (const float*)d_in[4];
  const float* cb        = (const float*)d_in[5];
  const float* multW     = (const float*)d_in[6];
  const float* chargeW   = (const float*)d_in[7];
  const float* gbf_means = (const float*)d_in[8];
  const float* gbf_stds  = (const float*)d_in[9];
  const float* gbf_mul   = (const float*)d_in[10];
  const float* gbf_bias  = (const float*)d_in[11];
  const float* proj1_W   = (const float*)d_in[12];
  const float* proj1_b   = (const float*)d_in[13];
  const float* proj2_W   = (const float*)d_in[14];
  const float* proj2_b   = (const float*)d_in[15];
  const float* dir_W     = (const float*)d_in[16];
  const float* dir_b     = (const float*)d_in[17];

  float* out      = (float*)d_out;
  float* node_out = out;                       // (4,256,256)
  float* gbf_out  = out + 262144;              // (4,256,256,256)
  float* dir_out  = out + 262144 + 67108864;   // (4,256,256,256)

  node_kernel<<<1024, 256, 0, stream>>>(an, emb_z, ec, cW, cb, multW, chargeW,
                                        node_out);
  pair_kernel<<<512, 256, SMEM_BYTES, stream>>>(
      positions, an, gbf_means, gbf_stds, gbf_mul, gbf_bias, proj1_W, proj1_b,
      proj2_W, proj2_b, dir_W, dir_b, gbf_out, dir_out);
}
